// GlobalTransformer_43258910605779
// MI455X (gfx1250) — matrix operand order repaired and checked by reading, not compiled, not run on hardware
//
#include <hip/hip_runtime.h>
#include <hip/hip_bf16.h>
#include <math.h>

typedef __attribute__((ext_vector_type(16))) _Float16 v16h;
typedef __attribute__((ext_vector_type(8)))  _Float16 v8h;
typedef __attribute__((ext_vector_type(8)))  float    v8f;

#define B_   2
#define T_   16
#define N_   256
#define H_   64
#define S_   (T_*N_)     // 4096
#define BS_  (B_*S_)     // 8192
#define DH_  16
#define MID_ 32
#define LOW_ 16

static __device__ __forceinline__ v8f wmma_f16(v16h a, v16h b, v8f c) {
  // D = A(16x32 f16) x B(32x16 f16) + C(16x16 f32)
  return __builtin_amdgcn_wmma_f32_16x16x32_f16(false, a, false, b, (short)0, c,
                                                false, false);
}

static __device__ __forceinline__ void cvt8(v16h& d, int off, float4 a, float4 b,
                                            float scl) {
  d[off + 0] = (_Float16)(a.x * scl); d[off + 1] = (_Float16)(a.y * scl);
  d[off + 2] = (_Float16)(a.z * scl); d[off + 3] = (_Float16)(a.w * scl);
  d[off + 4] = (_Float16)(b.x * scl); d[off + 5] = (_Float16)(b.y * scl);
  d[off + 6] = (_Float16)(b.z * scl); d[off + 7] = (_Float16)(b.w * scl);
}

// DPP lane exchange (VALU, no LDS) -- CTRL: 0xB1 xor1, 0x4E xor2,
// 0x124 row_ror:4, 0x128 row_ror:8
template <int CTRL>
static __device__ __forceinline__ float dppf(float v) {
  return __int_as_float(__builtin_amdgcn_update_dpp(
      __float_as_int(v), __float_as_int(v), CTRL, 0xf, 0xf, false));
}
// reduction across each 16-lane row (both wave halves independently)
static __device__ __forceinline__ float rmax16(float x) {
  x = fmaxf(x, dppf<0xB1>(x));
  x = fmaxf(x, dppf<0x4E>(x));
  x = fmaxf(x, dppf<0x124>(x));
  x = fmaxf(x, dppf<0x128>(x));
  return x;
}
static __device__ __forceinline__ float radd16(float x) {
  x += dppf<0xB1>(x);
  x += dppf<0x4E>(x);
  x += dppf<0x124>(x);
  x += dppf<0x128>(x);
  return x;
}

// ---------------------------------------------------------------------------
// x[b,s,h] = x_seq[b,s]*Wi[h] + bi[h] + sinusoidal_pos(t, h)
// ---------------------------------------------------------------------------
__global__ void k_input(const float* __restrict__ xseq, const float* __restrict__ Wi,
                        const float* __restrict__ bi, float* __restrict__ x) {
  int idx = blockIdx.x * blockDim.x + threadIdx.x;   // over BS_*H_
  if (idx >= BS_ * H_) return;
  int h   = idx & (H_ - 1);
  int row = idx >> 6;          // b*S + s
  int s   = row & (S_ - 1);
  int t   = s >> 8;            // s / 256
  float v = xseq[row] * Wi[h] + bi[h];
  int j = h >> 1;
  float freq = __expf(-(float)(2 * j) * (logf(10000.0f) / (float)H_));
  float ang  = (float)t * freq;
  v += (h & 1) ? __cosf(ang) : __sinf(ang);
  x[idx] = v;
}

// ---------------------------------------------------------------------------
// Weight pre-pass: Wt[n*K + k] = (f16) W[k*N + n]   (transpose + convert once)
// ---------------------------------------------------------------------------
__global__ void k_wt(const float* __restrict__ W, _Float16* __restrict__ Wt,
                     int K, int N) {
  int idx = blockIdx.x * blockDim.x + threadIdx.x;
  if (idx >= K * N) return;
  int n = idx / K, k = idx - n * K;
  Wt[idx] = (_Float16)W[(size_t)k * N + n];
}

// ---------------------------------------------------------------------------
// Y[M,N] = X[M,K] @ W[K,N] + bias[N] (optional ReLU).
// X fp32 (vectorized b128 loads + cvt), W pre-transposed f16 (contiguous
// 32B per B-fragment lane). One wave computes a 16x64 strip (4 tiles sharing
// one A fragment). K is a template constant -> K-loop fully unrolled.
// ---------------------------------------------------------------------------
template <int K>
__global__ void k_gemm(const float* __restrict__ X, const _Float16* __restrict__ Wt,
                       const float* __restrict__ bias, float* __restrict__ Y,
                       int M, int N, int relu) {
  int wave = (blockIdx.x * blockDim.x + threadIdx.x) >> 5;
  int lane = threadIdx.x & 31;
  int ngrp = N >> 6;                       // 64-column groups
  int mt   = M >> 4;
  if (wave >= mt * ngrp) return;           // wave-uniform exit, EXEC stays full
  int tm = wave / ngrp, tg = wave % ngrp;
  int m  = lane & 15, hi = lane >> 4;

  v8f acc[4] = {{}, {}, {}, {}};
#pragma unroll
  for (int kk = 0; kk < K; kk += 32) {
    // A fragment: per-lane rows are 16 contiguous fp32
    const float* arow = X + (size_t)(tm * 16 + m) * K + kk + (hi << 4);
    float4 a0 = *(const float4*)(arow);
    float4 a1 = *(const float4*)(arow + 4);
    float4 a2 = *(const float4*)(arow + 8);
    float4 a3 = *(const float4*)(arow + 12);
    v16h a;
    cvt8(a, 0, a0, a1, 1.0f);
    cvt8(a, 8, a2, a3, 1.0f);
#pragma unroll
    for (int c = 0; c < 4; ++c) {
      // B fragment: 16 contiguous f16 per lane from transposed weights
      const v16h* bp = (const v16h*)(Wt + (size_t)(tg * 64 + c * 16 + m) * K +
                                     kk + (hi << 4));
      acc[c] = wmma_f16(a, *bp, acc[c]);
    }
  }
#pragma unroll
  for (int c = 0; c < 4; ++c) {
    int n = tg * 64 + c * 16 + m;
    float bv = bias[n];
#pragma unroll
    for (int v = 0; v < 8; ++v) {
      float r = acc[c][v] + bv;
      if (relu) r = fmaxf(r, 0.0f);
      Y[(size_t)(tm * 16 + v + (hi << 3)) * N + n] = r;
    }
  }
}

// ---------------------------------------------------------------------------
// Flash attention, one wave per (b, head, 16-query tile). Block-causal mask is
// tile-exact (n=256 per timestep, multiple of 16): q-tile at timestep t
// attends key tiles [0, (t+1)*16), no per-element masking.
// K/V tiles stream through double-buffered, zero-padded LDS tiles via async
// DMA (global_load_async_to_lds_b128, ASYNCcnt): tile j+1 is in flight while
// tile j is computed. Softmax row max/sum use DPP butterflies (no LDS).
// LDS ops from one wave execute in order, so Pst store->load only needs a
// DScnt fence; async-vs-ds WAR is fenced by s_wait_dscnt before each issue.
// ---------------------------------------------------------------------------
__global__ void k_attn(const float* __restrict__ qkv, float* __restrict__ out) {
  __shared__ float    Kls[2][16 * 32];  // [key 0..15][ch 0..31], ch 16..31 = 0
  __shared__ float    Vls[2][32 * 16];  // [key 0..31][ch 0..15], key 16..31 = 0
  __shared__ _Float16 Pst[16 * 16];
  int qt = blockIdx.x;           // 0..255
  int h  = blockIdx.y;           // 0..3
  int b  = blockIdx.z;           // 0..1
  int lane = threadIdx.x;        // 0..31 (wave32)
  int m = lane & 15, hi = lane >> 4;
  int nk = ((qt >> 4) + 1) << 4;            // key tiles to visit
  const float* base = qkv + (size_t)b * S_ * 192;

  // zero the pad regions once (K channels 16..31, V keys 16..31, both buffers)
#pragma unroll
  for (int i = 0; i < 8; ++i) {
    int idx = i * 32 + lane;                     // 0..255
    int kpad = (idx >> 4) * 32 + 16 + (idx & 15);
    Kls[0][kpad] = 0.0f;       Kls[1][kpad] = 0.0f;
    Vls[0][256 + idx] = 0.0f;  Vls[1][256 + idx] = 0.0f;
  }

  // Q fragment (A layout), pre-scaled by 1/sqrt(dh)=0.25, K-dim padded to 32
  v16h qa;
  {
    const float* qrow = base + (size_t)(qt * 16 + m) * 192 + h * DH_ + (hi << 3);
    float4 q0 = *(const float4*)(qrow);
    float4 q1 = *(const float4*)(qrow + 4);
    cvt8(qa, 0, q0, q1, 0.25f);
#pragma unroll
    for (int e = 8; e < 16; ++e) qa[e] = (_Float16)0.0f;
  }

  v8f acc = {};
  float rowMax[8], rowSum[8];
#pragma unroll
  for (int v = 0; v < 8; ++v) { rowMax[v] = -1e30f; rowSum[v] = 0.0f; }

  // async staging geometry: 32 lanes x 16B per op = 8 tile rows per op
  int r  = lane >> 2;            // 0..7
  int c4 = (lane & 3) << 2;      // 0,4,8,12

  auto issue = [&](int j) {
    int bi = j & 1;
    const float* kt = base + (size_t)(j * 16) * 192 + 64 + h * DH_;
    const float* vt = kt + 64;
    unsigned kd0 = (unsigned)(unsigned long long)(void*)&Kls[bi][r * 32 + c4];
    unsigned kd1 = (unsigned)(unsigned long long)(void*)&Kls[bi][(r + 8) * 32 + c4];
    unsigned vd0 = (unsigned)(unsigned long long)(void*)&Vls[bi][r * 16 + c4];
    unsigned vd1 = (unsigned)(unsigned long long)(void*)&Vls[bi][(r + 8) * 16 + c4];
    unsigned long long ka0 = (unsigned long long)(const void*)(kt + r * 192 + c4);
    unsigned long long ka1 = (unsigned long long)(const void*)(kt + (r + 8) * 192 + c4);
    unsigned long long va0 = (unsigned long long)(const void*)(vt + r * 192 + c4);
    unsigned long long va1 = (unsigned long long)(const void*)(vt + (r + 8) * 192 + c4);
    asm volatile("global_load_async_to_lds_b128 %0, %1, off"
                 :: "v"(kd0), "v"(ka0) : "memory");
    asm volatile("global_load_async_to_lds_b128 %0, %1, off"
                 :: "v"(kd1), "v"(ka1) : "memory");
    asm volatile("global_load_async_to_lds_b128 %0, %1, off"
                 :: "v"(vd0), "v"(va0) : "memory");
    asm volatile("global_load_async_to_lds_b128 %0, %1, off"
                 :: "v"(vd1), "v"(va1) : "memory");
  };

  issue(0);
  for (int j = 0; j < nk; ++j) {
    int bi = j & 1;
    if (j + 1 < nk) {
      // prior ds reads of the other buffer must retire before DMA overwrites
      asm volatile("s_wait_dscnt 0x0" ::: "memory");
      issue(j + 1);
      // async loads complete in order: <=4 outstanding => tile j has landed
      asm volatile("s_wait_asynccnt 0x4" ::: "memory");
    } else {
      asm volatile("s_wait_asynccnt 0x0" ::: "memory");
    }

    // K^T fragment (B layout): col = key m, k-dim = channel hi*8+e (e<8),
    // 16+hi*8+(e-8) (e>=8, zero pad) -> ds b128 x4
    v16h kb;
    {
      const float* kp = &Kls[bi][m * 32 + (hi << 3)];
      float4 k0 = *(const float4*)(kp);
      float4 k1 = *(const float4*)(kp + 4);
      float4 k2 = *(const float4*)(kp + 16);
      float4 k3 = *(const float4*)(kp + 20);
      cvt8(kb, 0, k0, k1, 1.0f);
      cvt8(kb, 8, k2, k3, 1.0f);
    }
    v8f zc = {};
    v8f s = wmma_f16(qa, kb, zc);   // s[v] = scores[row v+8*hi][key col m]

    float p[8];
#pragma unroll
    for (int v = 0; v < 8; ++v) {
      float nm = fmaxf(rowMax[v], rmax16(s[v]));
      float sc = __expf(rowMax[v] - nm);
      float pv = __expf(s[v] - nm);
      rowSum[v] = rowSum[v] * sc + radd16(pv);
      acc[v]   *= sc;
      rowMax[v] = nm;
      p[v] = pv;
    }

    // Re-stage P (C layout) through LDS into A layout for the PV matmul
#pragma unroll
    for (int v = 0; v < 8; ++v)
      Pst[(v + (hi << 3)) * 16 + m] = (_Float16)p[v];
    asm volatile("s_wait_dscnt 0x0" ::: "memory");

    // P fragment: k<16 <=> e<8, 8 contiguous halves per lane -> one 16B ds load
    v16h pa;
    {
      v8h p8 = *(const v8h*)&Pst[m * 16 + (hi << 3)];
#pragma unroll
      for (int e = 0; e < 8; ++e) { pa[e] = p8[e]; pa[e + 8] = (_Float16)0.0f; }
    }
    // V fragment (B layout): col = ch m, k-dim = key hi*8+e (e<8), 16+hi*8+(e-8) (e>=8, pad rows are 0)
    v16h vb;
#pragma unroll
    for (int e = 0; e < 16; ++e)
      vb[e] = (_Float16)Vls[bi][((hi << 3) + e + ((e >> 3) << 3)) * 16 + m];

    acc = wmma_f16(pa, vb, acc);
  }

#pragma unroll
  for (int v = 0; v < 8; ++v) {
    int row = qt * 16 + v + (hi << 3);
    out[((size_t)b * S_ + row) * H_ + h * DH_ + m] = acc[v] / rowSum[v];
  }
}

// ---------------------------------------------------------------------------
// x = LayerNorm(x + y) * g + b, one wave per 64-element row
// ---------------------------------------------------------------------------
__global__ void k_addln(float* __restrict__ x, const float* __restrict__ y,
                        const float* __restrict__ g, const float* __restrict__ bt) {
  int row  = blockIdx.x;
  int lane = threadIdx.x;      // 0..31
  size_t o = (size_t)row * H_;
  float v0 = x[o + lane]      + y[o + lane];
  float v1 = x[o + lane + 32] + y[o + lane + 32];
  float s = v0 + v1, sq = v0 * v0 + v1 * v1;
#pragma unroll
  for (int off = 16; off > 0; off >>= 1) {
    s  += __shfl_xor(s,  off, 32);
    sq += __shfl_xor(sq, off, 32);
  }
  float mean = s * (1.0f / H_);
  float var  = sq * (1.0f / H_) - mean * mean;
  float inv  = rsqrtf(var + 1e-5f);
  x[o + lane]      = (v0 - mean) * inv * g[lane]      + bt[lane];
  x[o + lane + 32] = (v1 - mean) * inv * g[lane + 32] + bt[lane + 32];
}

// ---------------------------------------------------------------------------
// Decoder head on last timestep: 64->32 relu ->16 relu ->1  (tiny, scalar)
// ---------------------------------------------------------------------------
__global__ void k_dec(const float* __restrict__ x, const float* __restrict__ dW1,
                      const float* __restrict__ db1, const float* __restrict__ dW2,
                      const float* __restrict__ db2, const float* __restrict__ dW3,
                      const float* __restrict__ db3, float* __restrict__ out) {
  int idx = blockIdx.x * blockDim.x + threadIdx.x;   // b*N + n
  if (idx >= B_ * N_) return;
  int b = idx >> 8;
  int n = idx & 255;
  const float* row = x + ((size_t)b * S_ + (size_t)(T_ - 1) * N_ + n) * H_;
  float h1[MID_];
  for (int o = 0; o < MID_; ++o) {
    float a = db1[o];
    for (int k = 0; k < H_; ++k) a += row[k] * dW1[k * MID_ + o];
    h1[o] = fmaxf(a, 0.0f);
  }
  float h2[LOW_];
  for (int o = 0; o < LOW_; ++o) {
    float a = db2[o];
    for (int k = 0; k < MID_; ++k) a += h1[k] * dW2[k * LOW_ + o];
    h2[o] = fmaxf(a, 0.0f);
  }
  float a = db3[0];
  for (int k = 0; k < LOW_; ++k) a += h2[k] * dW3[k];
  out[idx] = a;
}

// ---------------------------------------------------------------------------
extern "C" void kernel_launch(void* const* d_in, const int* in_sizes, int n_in,
                              void* d_out, int out_size, void* d_ws, size_t ws_size,
                              hipStream_t stream) {
  (void)in_sizes; (void)n_in; (void)out_size; (void)ws_size;
  const float* xseq = (const float*)d_in[0];
  /* d_in[1] = edge_index (int), unused by reference */
  const float* Wi   = (const float*)d_in[2];
  const float* bi   = (const float*)d_in[3];
  const float* Wqkv = (const float*)d_in[4];
  const float* bqkv = (const float*)d_in[5];
  const float* Wo   = (const float*)d_in[6];
  const float* bo   = (const float*)d_in[7];
  const float* ln1g = (const float*)d_in[8];
  const float* ln1b = (const float*)d_in[9];
  const float* W1   = (const float*)d_in[10];
  const float* b1   = (const float*)d_in[11];
  const float* W2   = (const float*)d_in[12];
  const float* b2   = (const float*)d_in[13];
  const float* ln2g = (const float*)d_in[14];
  const float* ln2b = (const float*)d_in[15];
  const float* dW1  = (const float*)d_in[16];
  const float* db1  = (const float*)d_in[17];
  const float* dW2  = (const float*)d_in[18];
  const float* db2  = (const float*)d_in[19];
  const float* dW3  = (const float*)d_in[20];
  const float* db3  = (const float*)d_in[21];

  float* ws   = (float*)d_ws;
  float* x    = ws;                            // BS*64
  float* qkv  = x    + (size_t)BS_ * H_;       // BS*192
  float* attn = qkv  + (size_t)BS_ * 192;      // BS*64
  float* tmp  = attn + (size_t)BS_ * H_;       // BS*64
  float* f1   = tmp  + (size_t)BS_ * H_;       // BS*256
  _Float16* wt = (_Float16*)(f1 + (size_t)BS_ * 256);  // f16 transposed weights
  // per-layer f16 weight slabs (elements): qkv_t 12288 | wo_t 4096 | w1_t 16384 | w2_t 16384
  const size_t LSTRIDE = 49152;

  k_input<<<(BS_ * H_ + 255) / 256, 256, 0, stream>>>(xseq, Wi, bi, x);

  // one-time weight transpose+convert for both layers
  for (int l = 0; l < 2; ++l) {
    _Float16* wl = wt + (size_t)l * LSTRIDE;
    k_wt<<<(64 * 192 + 255) / 256, 256, 0, stream>>>(Wqkv + (size_t)l * 64 * 192,
                                                     wl, 64, 192);
    k_wt<<<(64 * 64 + 255) / 256, 256, 0, stream>>>(Wo + (size_t)l * 64 * 64,
                                                    wl + 12288, 64, 64);
    k_wt<<<(64 * 256 + 255) / 256, 256, 0, stream>>>(W1 + (size_t)l * 64 * 256,
                                                     wl + 16384, 64, 256);
    k_wt<<<(256 * 64 + 255) / 256, 256, 0, stream>>>(W2 + (size_t)l * 256 * 64,
                                                     wl + 32768, 256, 64);
  }

  for (int l = 0; l < 2; ++l) {
    _Float16* wl = wt + (size_t)l * LSTRIDE;
    int waves;
    // QKV projection: [BS,64] @ [64,192]
    waves = (BS_ / 16) * (192 / 64);
    k_gemm<64><<<(waves + 3) / 4, 128, 0, stream>>>(x, wl, bqkv + l * 192, qkv,
                                                    BS_, 192, 0);
    dim3 ga(S_ / 16, 4, B_);
    k_attn<<<ga, 32, 0, stream>>>(qkv, attn);

    // Output projection: [BS,64] @ [64,64]
    waves = (BS_ / 16) * (64 / 64);
    k_gemm<64><<<(waves + 3) / 4, 128, 0, stream>>>(attn, wl + 12288, bo + l * H_,
                                                    tmp, BS_, 64, 0);
    k_addln<<<BS_, 32, 0, stream>>>(x, tmp, ln1g + l * H_, ln1b + l * H_);

    // FFN1: [BS,64] @ [64,256] + ReLU
    waves = (BS_ / 16) * (256 / 64);
    k_gemm<64><<<(waves + 3) / 4, 128, 0, stream>>>(x, wl + 16384, b1 + l * 256,
                                                    f1, BS_, 256, 1);
    // FFN2: [BS,256] @ [256,64]
    waves = (BS_ / 16) * (64 / 64);
    k_gemm<256><<<(waves + 3) / 4, 128, 0, stream>>>(f1, wl + 32768, b2 + l * H_,
                                                     tmp, BS_, 64, 0);
    k_addln<<<BS_, 32, 0, stream>>>(x, tmp, ln2g + l * H_, ln2b + l * H_);
  }

  k_dec<<<(B_ * N_ + 63) / 64, 64, 0, stream>>>(x, dW1, db1, dW2, db2, dW3, db3,
                                                (float*)d_out);
}
